// LinearAttention_3015067042152
// MI455X (gfx1250) — matrix operand order repaired and checked by reading, not compiled, not run on hardware
//
#include <hip/hip_runtime.h>

// ---------------------------------------------------------------------------
// LinearAttention on MI455X (gfx1250, wave32, WMMA bf16 16x16x32, f32 accum)
//
// Pipeline (all on `stream`):
//  0. cast_weights_kernel   : w_qkv / w_proj fp32 -> bf16
//  1. transpose_cast_kernel : x (b,c,n) fp32 -> xt (b,n,c) bf16 (LDS tiled)
//  2. zero_kernel           : zero context / q-col stats
//  3. qkv_gemm_kernel       : qkv = xt @ Wqkv^T   (bf16 WMMA, async-to-LDS
//                             double-buffered tile staging)
//  4. qmax_kernel           : colmax_d = max_n q[b,h,n,d] (encoded atomic max)
//  5. qsum_kernel           : colsum_d = sum_n exp(q - colmax)
//  6. context_kernel        : ctx = softmax_row(k)^T @ v  (bf16 WMMA + atomics)
//  7. attn_out_kernel       : mixed = softmax_col(q) @ ctx, axis-mixed, bf16
//  8. proj_gemm_kernel      : out^T = Wproj @ mixed^T + bias (bf16 WMMA, async
//                             staging; output transpose free from C layout)
// ---------------------------------------------------------------------------

typedef __bf16 bf16;
typedef __attribute__((ext_vector_type(16))) bf16  v16bf;
typedef __attribute__((ext_vector_type(8)))  bf16  v8bf;
typedef __attribute__((ext_vector_type(8)))  float v8f;

#define SEQ   16384   // n = 128*128
#define CDIM  256
#define QKV_N 768
#define NHEAD 8
#define DHEAD 32
#define BATCH 8
#define ROWS  (BATCH * SEQ)   // 131072

union FragBF { v16bf v; v8bf h[2]; };

// A-operand (16x32, 16-bit): lanes 0-15 hold K 0..7 (v0-3) and 16..23 (v4-7);
// lanes 16-31 hold K 8..15 and 24..31.  LDS row-major [m][k], stride in elems.
__device__ inline v16bf load_frag_a(const bf16* base, int lane, int stride) {
  FragBF f;
  const int m = lane & 15, half = (lane >> 4) & 1;
  const bf16* p = base + m * stride + half * 8;
  f.h[0] = *(const v8bf*)(p);
  f.h[1] = *(const v8bf*)(p + 16);
  return f.v;
}

// B-operand (32x16, 16-bit): lanes 0-15 hold K 0..7 and 16..23; lanes 16-31 hold K 8..15 and 24..31.
// LDS stores B^T row-major [n][k], stride in elems.
__device__ inline v16bf load_frag_b(const bf16* base, int lane, int stride) {
  FragBF f;
  const int nidx = lane & 15, half = (lane >> 4) & 1;
  const bf16* p = base + nidx * stride + half * 8;
  f.h[0] = *(const v8bf*)(p);
  f.h[1] = *(const v8bf*)(p + 16);
  return f.v;
}

__device__ inline v8f wmma_bf16(v16bf a, v16bf b, v8f c) {
  return __builtin_amdgcn_wmma_f32_16x16x32_bf16(false, a, false, b,
                                                 (short)0, c, false, false);
}

// Async global->LDS copy of one 16B chunk per lane (CDNA5 ASYNCcnt path).
// Generic pointers to __shared__ carry the LDS byte offset in addr[31:0]
// (ISA 10.2 aperture rule), so the low dword is the dsaddr VGPR.
__device__ inline void async_b128(const bf16* g, bf16* l) {
  asm volatile("global_load_async_to_lds_b128 %0, %1, off"
               :: "v"((unsigned)(uintptr_t)l), "v"(g) : "memory");
}
__device__ inline void wait_async0() {
  asm volatile("s_wait_asynccnt 0" ::: "memory");
}

// Monotone float <-> uint mapping for atomic max on floats.
__device__ inline unsigned enc_f32(float f) {
  unsigned u = __float_as_uint(f);
  return (u >> 31) ? ~u : (u | 0x80000000u);
}
__device__ inline float dec_f32(unsigned u) {
  return (u >> 31) ? __uint_as_float(u & 0x7fffffffu) : __uint_as_float(~u);
}

// ---------------------------------------------------------------------------
__global__ void __launch_bounds__(256)
cast_weights_kernel(const float* __restrict__ wq, const float* __restrict__ wp,
                    bf16* __restrict__ wqb, bf16* __restrict__ wpb) {
  const int i = blockIdx.x * 256 + threadIdx.x;
  if (i < QKV_N * CDIM) wqb[i] = (bf16)wq[i];
  if (i < CDIM * CDIM)  wpb[i] = (bf16)wp[i];
}

// x (b,c,n) fp32 -> xt (b*n, c) bf16, 64x64 LDS tiles, coalesced both sides.
__global__ void __launch_bounds__(256)
transpose_cast_kernel(const float* __restrict__ x, bf16* __restrict__ xt) {
  __shared__ bf16 tile[64 * 72];
  const int b = blockIdx.z;
  const int n0 = blockIdx.x * 64;
  const int c0 = blockIdx.y * 64;
  const int t = threadIdx.x;
  const float* xb = x + (size_t)b * CDIM * SEQ;
  #pragma unroll
  for (int p = 0; p < 4; ++p) {
    const int lin = t + p * 256;        // 0..1023
    const int cc = lin >> 4;            // 0..63
    const int sg = lin & 15;            // nn = sg*4 .. +3
    const float4 vv = *(const float4*)(xb + (size_t)(c0 + cc) * SEQ + n0 + sg * 4);
    tile[(sg * 4 + 0) * 72 + cc] = (bf16)vv.x;
    tile[(sg * 4 + 1) * 72 + cc] = (bf16)vv.y;
    tile[(sg * 4 + 2) * 72 + cc] = (bf16)vv.z;
    tile[(sg * 4 + 3) * 72 + cc] = (bf16)vv.w;
  }
  __syncthreads();
  bf16* outb = xt + ((size_t)b * SEQ + n0) * CDIM + c0;
  #pragma unroll
  for (int p = 0; p < 2; ++p) {
    const int lin = t + p * 256;        // 0..511
    const int nn = lin >> 3;            // 0..63
    const int sg = lin & 7;             // 0..7
    *(v8bf*)(outb + (size_t)nn * CDIM + sg * 8) = *(const v8bf*)&tile[nn * 72 + sg * 8];
  }
}

__global__ void __launch_bounds__(256)
zero_kernel(float* __restrict__ ctx, unsigned* __restrict__ colmax,
            float* __restrict__ colsum) {
  const int i = blockIdx.x * 256 + threadIdx.x;
  if (i < 65536) ctx[i] = 0.0f;
  if (i < 2048) { colmax[i] = 0u; colsum[i] = 0.0f; }
}

// ---------------------------------------------------------------------------
// qkv[row][j] = sum_k xt[row][k] * wqb[j][k].  Tile 128x128x32, 8 waves (4x2),
// async-to-LDS double-buffered staging.
__global__ void __launch_bounds__(256)
qkv_gemm_kernel(const bf16* __restrict__ xt, const bf16* __restrict__ wqb,
                float* __restrict__ qkv) {
  __shared__ bf16 As[2][128 * 40];   // [i_local][k_local], pad 32->40
  __shared__ bf16 Bs[2][128 * 40];   // [j_local][k_local]
  const int t = threadIdx.x, lane = t & 31, wid = t >> 5;
  const int wm = wid >> 1, wn = wid & 1;
  const int rowTile = blockIdx.x * 128;   // flattened b*SEQ + n
  const int colTile = blockIdx.y * 128;   // 0..767

  v8f acc[2][4] = {};

  // prologue: stage k0 = 0 into buffer 0
  #pragma unroll
  for (int p = 0; p < 2; ++p) {
    const int lin = t + p * 256;
    const int row = lin >> 2, seg = lin & 3;   // seg*8 elems = 16B chunks
    async_b128(xt  + (size_t)(rowTile + row) * CDIM + seg * 8, &As[0][row * 40 + seg * 8]);
    async_b128(wqb + (size_t)(colTile + row) * CDIM + seg * 8, &Bs[0][row * 40 + seg * 8]);
  }

  int buf = 0;
  for (int k0 = 0; k0 < CDIM; k0 += 32) {
    wait_async0();
    __syncthreads();
    if (k0 + 32 < CDIM) {
      const int nb = buf ^ 1;
      #pragma unroll
      for (int p = 0; p < 2; ++p) {
        const int lin = t + p * 256;
        const int row = lin >> 2, seg = lin & 3;
        async_b128(xt  + (size_t)(rowTile + row) * CDIM + (k0 + 32) + seg * 8,
                   &As[nb][row * 40 + seg * 8]);
        async_b128(wqb + (size_t)(colTile + row) * CDIM + (k0 + 32) + seg * 8,
                   &Bs[nb][row * 40 + seg * 8]);
      }
    }
    v16bf af0 = load_frag_a(&As[buf][(wm * 32 + 0)  * 40], lane, 40);
    v16bf af1 = load_frag_a(&As[buf][(wm * 32 + 16) * 40], lane, 40);
    #pragma unroll
    for (int nt = 0; nt < 4; ++nt) {
      v16bf bfr = load_frag_b(&Bs[buf][(wn * 64 + nt * 16) * 40], lane, 40);
      acc[0][nt] = wmma_bf16(af0, bfr, acc[0][nt]);
      acc[1][nt] = wmma_bf16(af1, bfr, acc[1][nt]);
    }
    buf ^= 1;
  }

  const int half = lane >> 4, nloc = lane & 15;
  #pragma unroll
  for (int mt = 0; mt < 2; ++mt)
    #pragma unroll
    for (int nt = 0; nt < 4; ++nt)
      #pragma unroll
      for (int r = 0; r < 8; ++r) {
        const int row = rowTile + wm * 32 + mt * 16 + half * 8 + r;
        const int col = colTile + wn * 64 + nt * 16 + nloc;
        qkv[(size_t)row * QKV_N + col] = acc[mt][nt][r];
      }
}

// ---------------------------------------------------------------------------
// Column max of q over n, per (b,h,d).  grid = (16 splits, 64 bh)
__global__ void __launch_bounds__(256)
qmax_kernel(const float* __restrict__ qkv, unsigned* __restrict__ colmax) {
  __shared__ float red[256 * 33];
  const int bh = blockIdx.y, b = bh >> 3, h = bh & 7;
  const int t = threadIdx.x;
  const int n0 = blockIdx.x * (SEQ / 16);
  float m[DHEAD];
  #pragma unroll
  for (int d = 0; d < DHEAD; ++d) m[d] = -3.4e38f;
  const float* qb = qkv + (size_t)(b * SEQ) * QKV_N + h * DHEAD;
  for (int nn = n0 + t; nn < n0 + SEQ / 16; nn += 256) {
    const float* row = qb + (size_t)nn * QKV_N;
    #pragma unroll
    for (int d = 0; d < DHEAD; ++d) m[d] = fmaxf(m[d], row[d]);
  }
  #pragma unroll
  for (int d = 0; d < DHEAD; ++d) red[t * 33 + d] = m[d];
  __syncthreads();
  if (t < DHEAD) {
    float mm = -3.4e38f;
    for (int tt = 0; tt < 256; ++tt) mm = fmaxf(mm, red[tt * 33 + t]);
    atomicMax(&colmax[bh * DHEAD + t], enc_f32(mm));
  }
}

// Column sum of exp(q - colmax) over n.
__global__ void __launch_bounds__(256)
qsum_kernel(const float* __restrict__ qkv, const unsigned* __restrict__ colmax,
            float* __restrict__ colsum) {
  __shared__ float red[256 * 33];
  const int bh = blockIdx.y, b = bh >> 3, h = bh & 7;
  const int t = threadIdx.x;
  const int n0 = blockIdx.x * (SEQ / 16);
  float cm[DHEAD], s[DHEAD];
  #pragma unroll
  for (int d = 0; d < DHEAD; ++d) { cm[d] = dec_f32(colmax[bh * DHEAD + d]); s[d] = 0.0f; }
  const float* qb = qkv + (size_t)(b * SEQ) * QKV_N + h * DHEAD;
  for (int nn = n0 + t; nn < n0 + SEQ / 16; nn += 256) {
    const float* row = qb + (size_t)nn * QKV_N;
    #pragma unroll
    for (int d = 0; d < DHEAD; ++d) s[d] += __expf(row[d] - cm[d]);
  }
  #pragma unroll
  for (int d = 0; d < DHEAD; ++d) red[t * 33 + d] = s[d];
  __syncthreads();
  if (t < DHEAD) {
    float ss = 0.0f;
    for (int tt = 0; tt < 256; ++tt) ss += red[tt * 33 + t];
    atomicAdd(&colsum[bh * DHEAD + t], ss);
  }
}

// ---------------------------------------------------------------------------
// ctx[bh][d][e] = sum_n softmax_row(k)[n][d] * v[n][e]
// grid = (8 n-splits, 64 bh); 256 rows staged/iter; 8 waves = 4 tiles x 2
// K-halves; partials merged via global atomicAdd.
__global__ void __launch_bounds__(256)
context_kernel(const float* __restrict__ qkv, float* __restrict__ ctx) {
  __shared__ bf16 kT[32 * 264];   // [d][nn], pad 256->264
  __shared__ bf16 vT[32 * 264];   // [e][nn]
  const int bh = blockIdx.y, b = bh >> 3, h = bh & 7;
  const int t = threadIdx.x, lane = t & 31, wid = t >> 5;
  const int tile = wid & 3, khalf = wid >> 2;
  const int d0 = (tile >> 1) * 16, e0 = (tile & 1) * 16;
  const int n0 = blockIdx.x * (SEQ / 8);
  const float* kb = qkv + (size_t)(b * SEQ) * QKV_N + CDIM + h * DHEAD;
  const float* vb = qkv + (size_t)(b * SEQ) * QKV_N + 2 * CDIM + h * DHEAD;

  v8f acc = {};
  for (int it = 0; it < (SEQ / 8) / 256; ++it) {
    const int nn = n0 + it * 256 + t;
    const float* kr = kb + (size_t)nn * QKV_N;
    float kv[DHEAD], mx = -3.4e38f;
    #pragma unroll
    for (int d = 0; d < DHEAD; ++d) { kv[d] = kr[d]; mx = fmaxf(mx, kv[d]); }
    float s = 0.0f;
    #pragma unroll
    for (int d = 0; d < DHEAD; ++d) { kv[d] = __expf(kv[d] - mx); s += kv[d]; }
    const float inv = 1.0f / s;
    #pragma unroll
    for (int d = 0; d < DHEAD; ++d) kT[d * 264 + t] = (bf16)(kv[d] * inv);
    const float* vr = vb + (size_t)nn * QKV_N;
    #pragma unroll
    for (int e = 0; e < DHEAD; ++e) vT[e * 264 + t] = (bf16)vr[e];
    __syncthreads();
    #pragma unroll
    for (int kk = 0; kk < 4; ++kk) {
      const int ko = khalf * 128 + kk * 32;
      v16bf af  = load_frag_a(&kT[d0 * 264 + ko], lane, 264);
      v16bf bfr = load_frag_b(&vT[e0 * 264 + ko], lane, 264);
      acc = wmma_bf16(af, bfr, acc);
    }
    __syncthreads();
  }
  const int half = lane >> 4, nloc = lane & 15;
  #pragma unroll
  for (int r = 0; r < 8; ++r)
    atomicAdd(&ctx[bh * 1024 + (d0 + half * 8 + r) * DHEAD + e0 + nloc], acc[r]);
}

// ---------------------------------------------------------------------------
// mixed[b*SEQ + (h*2048 + nn/8)][(nn%8)*32 + e] = (query @ ctx)[nn][e]  (bf16)
__global__ void __launch_bounds__(256)
attn_out_kernel(const float* __restrict__ qkv, const float* __restrict__ ctx,
                const unsigned* __restrict__ colmax, const float* __restrict__ colsum,
                bf16* __restrict__ mixed) {
  __shared__ float c[1024];
  __shared__ float cm[DHEAD], cs[DHEAD];
  const int bh = blockIdx.y, b = bh >> 3, h = bh & 7;
  const int t = threadIdx.x;
  for (int i = t; i < 1024; i += 256) c[i] = ctx[bh * 1024 + i];
  if (t < DHEAD) {
    cm[t] = dec_f32(colmax[bh * DHEAD + t]);
    cs[t] = 1.0f / colsum[bh * DHEAD + t];
  }
  __syncthreads();
  const int nn = blockIdx.x * 256 + t;
  const float* qr = qkv + (size_t)(b * SEQ + nn) * QKV_N + h * DHEAD;
  float qv[DHEAD];
  #pragma unroll
  for (int d = 0; d < DHEAD; ++d) qv[d] = __expf(qr[d] - cm[d]) * cs[d];
  float o[DHEAD];
  #pragma unroll
  for (int e = 0; e < DHEAD; ++e) o[e] = 0.0f;
  #pragma unroll
  for (int d = 0; d < DHEAD; ++d) {
    const float qd = qv[d];
    #pragma unroll
    for (int e = 0; e < DHEAD; ++e) o[e] = fmaf(qd, c[d * DHEAD + e], o[e]);
  }
  const int i = h * 2048 + (nn >> 3);
  const int col = (nn & 7) * DHEAD;
  bf16* dst = mixed + (size_t)(b * SEQ + i) * CDIM + col;
  #pragma unroll
  for (int e = 0; e < DHEAD; ++e) dst[e] = (bf16)o[e];
}

// ---------------------------------------------------------------------------
// out[b][j][i] = sum_k mixed[b*SEQ+i][k] * w_proj[j][k] + b_proj[j].
// Computed as y^T = Wproj @ mixed^T -> coalesced stores in i.  Async staging.
__global__ void __launch_bounds__(256)
proj_gemm_kernel(const bf16* __restrict__ mixed, const bf16* __restrict__ wpb,
                 const float* __restrict__ b_proj, float* __restrict__ out) {
  __shared__ bf16 As[2][128 * 40];   // w_proj rows j (A: M=j, K=k)
  __shared__ bf16 Bs[2][128 * 40];   // mixed rows i (B^T row-major)
  const int t = threadIdx.x, lane = t & 31, wid = t >> 5;
  const int wm = wid >> 1, wn = wid & 1;
  const int iTile = blockIdx.x * 128;   // flattened b*SEQ + n
  const int jTile = blockIdx.y * 128;   // 0 or 128

  v8f acc[2][4] = {};

  #pragma unroll
  for (int p = 0; p < 2; ++p) {
    const int lin = t + p * 256;
    const int row = lin >> 2, seg = lin & 3;
    async_b128(wpb   + (size_t)(jTile + row) * CDIM + seg * 8, &As[0][row * 40 + seg * 8]);
    async_b128(mixed + (size_t)(iTile + row) * CDIM + seg * 8, &Bs[0][row * 40 + seg * 8]);
  }

  int buf = 0;
  for (int k0 = 0; k0 < CDIM; k0 += 32) {
    wait_async0();
    __syncthreads();
    if (k0 + 32 < CDIM) {
      const int nb = buf ^ 1;
      #pragma unroll
      for (int p = 0; p < 2; ++p) {
        const int lin = t + p * 256;
        const int row = lin >> 2, seg = lin & 3;
        async_b128(wpb   + (size_t)(jTile + row) * CDIM + (k0 + 32) + seg * 8,
                   &As[nb][row * 40 + seg * 8]);
        async_b128(mixed + (size_t)(iTile + row) * CDIM + (k0 + 32) + seg * 8,
                   &Bs[nb][row * 40 + seg * 8]);
      }
    }
    v16bf af0 = load_frag_a(&As[buf][(wm * 32 + 0)  * 40], lane, 40);
    v16bf af1 = load_frag_a(&As[buf][(wm * 32 + 16) * 40], lane, 40);
    #pragma unroll
    for (int nt = 0; nt < 4; ++nt) {
      v16bf bfr = load_frag_b(&Bs[buf][(wn * 64 + nt * 16) * 40], lane, 40);
      acc[0][nt] = wmma_bf16(af0, bfr, acc[0][nt]);
      acc[1][nt] = wmma_bf16(af1, bfr, acc[1][nt]);
    }
    buf ^= 1;
  }

  const int half = lane >> 4, nloc = lane & 15;
  #pragma unroll
  for (int mt = 0; mt < 2; ++mt)
    #pragma unroll
    for (int nt = 0; nt < 4; ++nt) {
      const int j0 = jTile + wm * 32 + mt * 16 + half * 8;
      const int iflat = iTile + wn * 64 + nt * 16 + nloc;
      const int b = iflat >> 14, iloc = iflat & (SEQ - 1);
      #pragma unroll
      for (int r = 0; r < 8; ++r)
        out[(size_t)(b * CDIM + j0 + r) * SEQ + iloc] = acc[mt][nt][r] + b_proj[j0 + r];
    }
}

// ---------------------------------------------------------------------------
extern "C" void kernel_launch(void* const* d_in, const int* in_sizes, int n_in,
                              void* d_out, int out_size, void* d_ws, size_t ws_size,
                              hipStream_t stream) {
  const float* x      = (const float*)d_in[0];   // (8,256,128,128)
  const float* w_qkv  = (const float*)d_in[1];   // (768,256)
  const float* w_proj = (const float*)d_in[2];   // (256,256)
  const float* b_proj = (const float*)d_in[3];   // (256,)
  float* out = (float*)d_out;

  char* ws = (char*)d_ws;
  size_t off = 0;
  float*    qkv    = (float*)(ws + off);    off += (size_t)ROWS * QKV_N * sizeof(float);   // 402.7 MB
  bf16*     mixed  = (bf16*)(ws + off);     off += (size_t)ROWS * CDIM * sizeof(bf16);     // 67.1 MB
  bf16*     xt     = (bf16*)(ws + off);     off += (size_t)ROWS * CDIM * sizeof(bf16);     // 67.1 MB
  bf16*     wqb    = (bf16*)(ws + off);     off += (size_t)QKV_N * CDIM * sizeof(bf16);
  bf16*     wpb    = (bf16*)(ws + off);     off += (size_t)CDIM * CDIM * sizeof(bf16);
  float*    ctx    = (float*)(ws + off);    off += (size_t)65536 * sizeof(float);
  unsigned* colmax = (unsigned*)(ws + off); off += (size_t)2048 * sizeof(unsigned);
  float*    colsum = (float*)(ws + off);    off += (size_t)2048 * sizeof(float);
  (void)ws_size; (void)in_sizes; (void)n_in; (void)out_size;

  cast_weights_kernel<<<QKV_N * CDIM / 256, 256, 0, stream>>>(w_qkv, w_proj, wqb, wpb);
  transpose_cast_kernel<<<dim3(SEQ / 64, CDIM / 64, BATCH), 256, 0, stream>>>(x, xt);
  zero_kernel<<<256, 256, 0, stream>>>(ctx, colmax, colsum);
  qkv_gemm_kernel<<<dim3(ROWS / 128, QKV_N / 128), 256, 0, stream>>>(xt, wqb, qkv);
  qmax_kernel<<<dim3(16, 64), 256, 0, stream>>>(qkv, colmax);
  qsum_kernel<<<dim3(16, 64), 256, 0, stream>>>(qkv, colmax, colsum);
  context_kernel<<<dim3(8, 64), 256, 0, stream>>>(qkv, ctx);
  attn_out_kernel<<<dim3(SEQ / 256, 64), 256, 0, stream>>>(qkv, ctx, colmax, colsum, mixed);
  proj_gemm_kernel<<<dim3(ROWS / 128, CDIM / 128), 256, 0, stream>>>(mixed, wpb, b_proj, out);
}
